// LinearAttention_5471788335538
// MI455X (gfx1250) — matrix operand order repaired and checked by reading, not compiled, not run on hardware
//
#include <hip/hip_runtime.h>
#include <hip/hip_bf16.h>
#include <math.h>

// ---------------- problem constants ----------------
#define B_     8
#define C_     256
#define HW_    128
#define N_     16384L          // HW_*HW_
#define INNER_ 512
#define HEADS_ 8
#define D_     64

// ---------------- types ----------------
typedef __bf16 bf16x16 __attribute__((ext_vector_type(16)));
typedef float  floatx8 __attribute__((ext_vector_type(8)));
typedef unsigned int uint32x4 __attribute__((ext_vector_type(4)));
typedef unsigned int uint32x2 __attribute__((ext_vector_type(2)));

union Frag16 {
    bf16x16  v;
    uint32x4 q[2];
    __bf16   h[16];
};

union Pack4 {                   // 4 bf16 <-> 8 bytes, for register-blocked transpose
    uint32x2 d;
    __bf16   h[4];
};

__device__ __forceinline__ floatx8 zero8() {
    floatx8 z = {0.f,0.f,0.f,0.f,0.f,0.f,0.f,0.f};
    return z;
}

__device__ __forceinline__ floatx8 wmma_bf16(bf16x16 a, bf16x16 b, floatx8 c) {
    // D = A(16x32 bf16) * B(32x16 bf16) + C(16x16 f32)
    return __builtin_amdgcn_wmma_f32_16x16x32_bf16(false, a, false, b,
                                                   (short)0, c, false, false);
}

// A fragment (16x32, MxK). Lanes 0-15 row M=l hold K 0..15,
// lanes 16-31 row M=l-16 hold K 16..31. Row-major tile, stride ldr
// elements (ldr*2 must be a multiple of 16 bytes).
__device__ __forceinline__ void load_a_frag(Frag16& f, const __bf16* rowbase,
                                            int lane, int ldr, int koff) {
    const int l = lane & 15, hi = lane >> 4;
    const uint32x4* p =
        reinterpret_cast<const uint32x4*>(rowbase + (long)l * ldr + koff);
    f.q[0] = p[2 * hi];       // K 0..7  / 16..23
    f.q[1] = p[2 * hi + 1];   // K 8..15 / 24..31
}

// B fragment (32x16, KxN). Lanes 0-15: col N=l, K 0..15; lanes 16-31: col N=l,
// K 16..31. LDS tile stored TRANSPOSED as [n][k] rows so each lane reads
// 32 contiguous bytes.
__device__ __forceinline__ void load_b_frag(Frag16& f, const __bf16* colbase,
                                            int lane, int ldr, int koff) {
    const int l = lane & 15, hi = lane >> 4;
    const uint32x4* p =
        reinterpret_cast<const uint32x4*>(colbase + (long)l * ldr + koff + hi * 16);
    f.q[0] = p[0];
    f.q[1] = p[1];
}

// Stage one 128x32 A tile (b128 loads/stores) and one 32x128 B tile
// (register-blocked 4x4 transpose: 4x b64 loads -> 4x b64 LDS stores).
__device__ __forceinline__ void stage_tiles(
    const __bf16* __restrict__ A, long lda,
    const __bf16* __restrict__ X, long ldx, int k0,
    __bf16 (*As)[40], __bf16 (*Bs)[40],
    int ar, int ah, int bkg, int bng)
{
    const uint32x4* ag = reinterpret_cast<const uint32x4*>(A + (long)ar * lda + k0);
    uint32x4 a0 = ag[ah * 2], a1 = ag[ah * 2 + 1];
    uint32x4* asr = reinterpret_cast<uint32x4*>(&As[ar][0]);
    asr[ah * 2]     = a0;
    asr[ah * 2 + 1] = a1;

    Pack4 r[4];
    #pragma unroll
    for (int i = 0; i < 4; ++i)
        r[i].d = *reinterpret_cast<const uint32x2*>(
                     X + (long)(k0 + bkg * 4 + i) * ldx + bng * 4);
    #pragma unroll
    for (int j = 0; j < 4; ++j) {
        Pack4 w;
        w.h[0] = r[0].h[j]; w.h[1] = r[1].h[j];
        w.h[2] = r[2].h[j]; w.h[3] = r[3].h[j];
        *reinterpret_cast<uint32x2*>(&Bs[bng * 4 + j][bkg * 4]) = w.d;
    }
}

// ---------------- shared 128x128 bf16 WMMA tile ----------------
// 8 waves (2x4 grid, 64x32 per wave). Double-buffered LDS: one barrier per
// 32-K step; step i+1 staged while WMMAs consume step i.
__device__ __forceinline__ void gemm_tile_128x128(
    const __bf16* __restrict__ A, long lda,    // weights [M][K] row-major, offset to mbase
    const __bf16* __restrict__ X, long ldx,    // activations [K][N], offset to n0
    int K, __bf16 (*As)[128][40], __bf16 (*Bs)[128][40], floatx8 acc[4][2])
{
    const int tid  = threadIdx.x;
    const int lane = tid & 31;
    const int wave = tid >> 5;
    const int wm = wave & 1;                  // 2 waves along M (64 rows each)
    const int wn = wave >> 1;                 // 4 waves along N (32 cols each)
    const int ar  = tid >> 1, ah  = tid & 1;  // A stage: row / 16-elem half
    const int bkg = tid >> 5, bng = tid & 31; // B stage: 4-k group / 4-n group

    stage_tiles(A, lda, X, ldx, 0, As[0], Bs[0], ar, ah, bkg, bng);
    int cur = 0;
    for (int k0 = 0; k0 < K; k0 += 32) {
        __syncthreads();
        if (k0 + 32 < K) {
            stage_tiles(A, lda, X, ldx, k0 + 32, As[cur ^ 1], Bs[cur ^ 1],
                        ar, ah, bkg, bng);
            __builtin_prefetch(A + (long)ar * lda + k0 + 64, 0, 1);
            __builtin_prefetch(X + (long)(k0 + 64 + bkg * 4) * ldx + bng * 4, 0, 1);
        }
        Frag16 af[4];
        #pragma unroll
        for (int mf = 0; mf < 4; ++mf)
            load_a_frag(af[mf], &As[cur][wm * 64 + mf * 16][0], lane, 40, 0);
        Frag16 bfr[2];
        #pragma unroll
        for (int nf = 0; nf < 2; ++nf)
            load_b_frag(bfr[nf], &Bs[cur][wn * 32 + nf * 16][0], lane, 40, 0);
        #pragma unroll
        for (int mf = 0; mf < 4; ++mf)
            #pragma unroll
            for (int nf = 0; nf < 2; ++nf)
                acc[mf][nf] = wmma_bf16(af[mf].v, bfr[nf].v, acc[mf][nf]);
        cur ^= 1;
    }
}

// ---------------- utility kernels ----------------
__global__ __launch_bounds__(256) void k_convert_bf16(const float* __restrict__ in,
                                                      __bf16* __restrict__ out, long n) {
    long i = (long)blockIdx.x * blockDim.x + threadIdx.x;
    long stride = (long)gridDim.x * blockDim.x;
    for (; i < n; i += stride) out[i] = (__bf16)in[i];
}

__global__ __launch_bounds__(256) void k_zero_f32(float* __restrict__ p, long n) {
    long i = (long)blockIdx.x * blockDim.x + threadIdx.x;
    long stride = (long)gridDim.x * blockDim.x;
    for (; i < n; i += stride) p[i] = 0.f;
}

// depthwise 3x3, pad 1, fp32 in -> bf16 out
__global__ __launch_bounds__(128) void k_dwconv(const float* __restrict__ fmap,
                                                const float* __restrict__ wdw,
                                                __bf16* __restrict__ out) {
    const int x  = threadIdx.x;          // 0..127
    const int y  = blockIdx.x & (HW_ - 1);
    const int bc = blockIdx.x >> 7;      // 0..B_*C_-1
    const int c  = bc & (C_ - 1);
    const float* w  = wdw + c * 9;
    const float* in = fmap + (long)bc * N_;
    float s = 0.f;
    #pragma unroll
    for (int dy = -1; dy <= 1; ++dy) {
        const int yy = y + dy;
        if (yy < 0 || yy >= HW_) continue;
        #pragma unroll
        for (int dx = -1; dx <= 1; ++dx) {
            const int xx = x + dx;
            if (xx < 0 || xx >= HW_) continue;
            s += w[(dy + 1) * 3 + (dx + 1)] * in[yy * HW_ + xx];
        }
    }
    out[(long)bc * N_ + y * HW_ + x] = (__bf16)s;
}

// ---------------- q projection + fused softmax over d (per head) ----------------
__global__ __launch_bounds__(256) void k_gemm_q(const __bf16* __restrict__ wq,
                                                const __bf16* __restrict__ xb,
                                                __bf16* __restrict__ qout) {
    __shared__ __align__(16) __bf16 As[2][128][40];
    __shared__ __align__(16) __bf16 Bs[2][128][40];
    const int n0    = blockIdx.x * 128;
    const int mbase = blockIdx.y * 128;   // rows of w_q (512 total, 2 heads/tile)
    const int b     = blockIdx.z;
    floatx8 acc[4][2];
    #pragma unroll
    for (int i = 0; i < 4; ++i) { acc[i][0] = zero8(); acc[i][1] = zero8(); }
    gemm_tile_128x128(wq + (long)mbase * C_, C_,
                      xb + (long)b * C_ * N_ + n0, N_, C_, As, Bs, acc);

    const int tid = threadIdx.x, lane = tid & 31, wave = tid >> 5;
    const int wm = wave & 1, wn = wave >> 1, l0 = lane & 15, hi = lane >> 4;
    const float scale = 0.125f;   // 64^-0.5
    // wave owns a full head (64 rows) x 32 cols -> softmax over d in-register,
    // combining the lane pair (l, l^16) that shares a column.
    #pragma unroll
    for (int nf = 0; nf < 2; ++nf) {
        float m = -3.0e38f;
        #pragma unroll
        for (int mf = 0; mf < 4; ++mf)
            #pragma unroll
            for (int r = 0; r < 8; ++r) m = fmaxf(m, acc[mf][nf][r]);
        m = fmaxf(m, __shfl_xor(m, 16));
        float ex[4][8];
        float s = 0.f;
        #pragma unroll
        for (int mf = 0; mf < 4; ++mf)
            #pragma unroll
            for (int r = 0; r < 8; ++r) {
                ex[mf][r] = expf(acc[mf][nf][r] - m);
                s += ex[mf][r];
            }
        s += __shfl_xor(s, 16);
        const float inv = scale / s;
        const long col = n0 + wn * 32 + nf * 16 + l0;
        #pragma unroll
        for (int mf = 0; mf < 4; ++mf)
            #pragma unroll
            for (int r = 0; r < 8; ++r) {
                const int row = mbase + wm * 64 + mf * 16 + hi * 8 + r;
                qout[((long)b * INNER_ + row) * N_ + col] = (__bf16)(ex[mf][r] * inv);
            }
    }
}

// ---------------- kv projection (1024 rows: k then v) ----------------
__global__ __launch_bounds__(256) void k_gemm_kv(const __bf16* __restrict__ wpw,
                                                 const __bf16* __restrict__ xb,
                                                 __bf16* __restrict__ kvout) {
    __shared__ __align__(16) __bf16 As[2][128][40];
    __shared__ __align__(16) __bf16 Bs[2][128][40];
    const int n0    = blockIdx.x * 128;
    const int mbase = blockIdx.y * 128;   // 0..1023
    const int b     = blockIdx.z;
    floatx8 acc[4][2];
    #pragma unroll
    for (int i = 0; i < 4; ++i) { acc[i][0] = zero8(); acc[i][1] = zero8(); }
    gemm_tile_128x128(wpw + (long)mbase * C_, C_,
                      xb + (long)b * C_ * N_ + n0, N_, C_, As, Bs, acc);
    const int tid = threadIdx.x, lane = tid & 31, wave = tid >> 5;
    const int wm = wave & 1, wn = wave >> 1, l0 = lane & 15, hi = lane >> 4;
    #pragma unroll
    for (int mf = 0; mf < 4; ++mf)
        #pragma unroll
        for (int nf = 0; nf < 2; ++nf) {
            const long col = n0 + wn * 32 + nf * 16 + l0;
            #pragma unroll
            for (int r = 0; r < 8; ++r) {
                const int row = mbase + wm * 64 + mf * 16 + hi * 8 + r;
                kvout[((long)b * 2 * INNER_ + row) * N_ + col] = (__bf16)acc[mf][nf][r];
            }
        }
}

// ---------------- per-(b,channel) max & sum(exp) over N for k-softmax ----------------
__global__ __launch_bounds__(256) void k_kstats(const __bf16* __restrict__ kv,
                                                float* __restrict__ kmax,
                                                float* __restrict__ ksum) {
    const int b   = blockIdx.x >> 9;       // /512
    const int row = blockIdx.x & 511;      // k channel within batch
    const __bf16* p = kv + ((long)b * 2 * INNER_ + row) * N_;
    __shared__ float red[256];
    const int t = threadIdx.x;
    float m = -3.0e38f;
    for (long i = t; i < N_; i += 256) m = fmaxf(m, (float)p[i]);
    red[t] = m; __syncthreads();
    for (int s = 128; s > 0; s >>= 1) {
        if (t < s) red[t] = fmaxf(red[t], red[t + s]);
        __syncthreads();
    }
    m = red[0]; __syncthreads();
    float sum = 0.f;
    for (long i = t; i < N_; i += 256) sum += expf((float)p[i] - m);
    red[t] = sum; __syncthreads();
    for (int s = 128; s > 0; s >>= 1) {
        if (t < s) red[t] += red[t + s];
        __syncthreads();
    }
    if (t == 0) { kmax[blockIdx.x] = m; ksum[blockIdx.x] = red[0]; }
}

// ---------------- context = softmax(k)^T v per (b,head): 64x64, split-K atomics ----------------
__global__ __launch_bounds__(256) void k_context(const __bf16* __restrict__ kv,
                                                 const float* __restrict__ kmax,
                                                 const float* __restrict__ ksum,
                                                 float* __restrict__ ctx) {
    const int chunk = blockIdx.x;    // 0..7, K slice of 2048
    const int bh    = blockIdx.y;    // 0..63
    const int b = bh >> 3, h = bh & 7;
    const __bf16* krows = kv + ((long)b * 2 * INNER_ + h * 64) * N_;
    const __bf16* vrows = kv + ((long)b * 2 * INNER_ + INNER_ + h * 64) * N_;
    const float* km = kmax + b * 512 + h * 64;
    const float* ks = ksum + b * 512 + h * 64;
    __shared__ __align__(16) __bf16 Ks[2][64][40];
    __shared__ __align__(16) __bf16 Vs[2][64][40];
    __shared__ float skm[64], srk[64];
    const int t = threadIdx.x;
    if (t < 64) { skm[t] = km[t]; srk[t] = 1.0f / ks[t]; }
    __syncthreads();
    const int lane = t & 31, wave = t >> 5;
    const int mf = wave & 3, nfb = (wave >> 2) * 2;   // wave -> 1 A-frag, 2 B-frags
    floatx8 acc0 = zero8(), acc1 = zero8();
    const long nb0 = (long)chunk * 2048;

    // double-buffered staging: one barrier per 32-column step
    const int sd = t >> 5, skk = (t & 31);   // 8 elems per thread per tile
    auto stage = [&](int s, int buf) {
        const long nb = nb0 + (long)s * 32;
        #pragma unroll
        for (int rep = 0; rep < 8; ++rep) {
            const int d = sd * 8 + rep;
            Ks[buf][d][skk] =
                (__bf16)(expf((float)krows[(long)d * N_ + nb + skk] - skm[d]) * srk[d]);
            Vs[buf][d][skk] = vrows[(long)d * N_ + nb + skk];
        }
    };
    stage(0, 0);
    int cur = 0;
    for (int s = 0; s < 64; ++s) {
        __syncthreads();
        if (s + 1 < 64) stage(s + 1, cur ^ 1);
        Frag16 a, b0, b1;
        load_a_frag(a,  &Ks[cur][mf * 16][0], lane, 40, 0);
        load_b_frag(b0, &Vs[cur][nfb * 16][0], lane, 40, 0);
        load_b_frag(b1, &Vs[cur][(nfb + 1) * 16][0], lane, 40, 0);
        acc0 = wmma_bf16(a.v, b0.v, acc0);
        acc1 = wmma_bf16(a.v, b1.v, acc1);
        cur ^= 1;
    }
    const int l0 = lane & 15, hi = lane >> 4;
    #pragma unroll
    for (int r = 0; r < 8; ++r) {
        const int d = mf * 16 + hi * 8 + r;
        atomicAdd(&ctx[((long)bh * 64 + d) * 64 + nfb * 16 + l0], acc0[r]);
        atomicAdd(&ctx[((long)bh * 64 + d) * 64 + (nfb + 1) * 16 + l0], acc1[r]);
    }
}

// ---------------- out = ctx^T * q, GELU(exact), store bf16 ----------------
__global__ __launch_bounds__(256) void k_out(const float* __restrict__ ctx,
                                             const __bf16* __restrict__ qb,
                                             __bf16* __restrict__ attn) {
    const int n0 = blockIdx.x * 256;
    const int bh = blockIdx.y;
    const int b = bh >> 3, h = bh & 7;
    __shared__ __align__(16) __bf16 As[64][72];    // ctx^T: A[e][d]
    __shared__ __align__(16) __bf16 Bs[256][72];   // q^T:  B[n][d]
    const int t = threadIdx.x;
    const float* c = ctx + (long)bh * 4096;
    for (int e = t; e < 4096; e += 256) {
        const int d = e >> 6, ee = e & 63;
        As[ee][d] = (__bf16)c[d * 64 + ee];
    }
    const __bf16* q0 = qb + ((long)b * INNER_ + h * 64) * N_ + n0;
    // register-blocked 4x4 transpose staging of the 64x256 q tile
    for (int slot = t; slot < 1024; slot += 256) {
        const int ng = slot & 63, kg = slot >> 6;   // ng: 4-col group, kg: 4-row group
        Pack4 r[4];
        #pragma unroll
        for (int i = 0; i < 4; ++i)
            r[i].d = *reinterpret_cast<const uint32x2*>(
                         q0 + (long)(kg * 4 + i) * N_ + ng * 4);
        #pragma unroll
        for (int j = 0; j < 4; ++j) {
            Pack4 w;
            w.h[0] = r[0].h[j]; w.h[1] = r[1].h[j];
            w.h[2] = r[2].h[j]; w.h[3] = r[3].h[j];
            *reinterpret_cast<uint32x2*>(&Bs[ng * 4 + j][kg * 4]) = w.d;
        }
    }
    __syncthreads();
    const int lane = t & 31, wave = t >> 5;   // 8 waves along N, 32 cols each
    floatx8 acc[4][2];
    #pragma unroll
    for (int i = 0; i < 4; ++i) { acc[i][0] = zero8(); acc[i][1] = zero8(); }
    #pragma unroll
    for (int kstep = 0; kstep < 2; ++kstep) {
        Frag16 af[4];
        #pragma unroll
        for (int mf = 0; mf < 4; ++mf)
            load_a_frag(af[mf], &As[mf * 16][0], lane, 72, kstep * 32);
        Frag16 bfr[2];
        #pragma unroll
        for (int nf = 0; nf < 2; ++nf)
            load_b_frag(bfr[nf], &Bs[wave * 32 + nf * 16][0], lane, 72, kstep * 32);
        #pragma unroll
        for (int mf = 0; mf < 4; ++mf)
            #pragma unroll
            for (int nf = 0; nf < 2; ++nf)
                acc[mf][nf] = wmma_bf16(af[mf].v, bfr[nf].v, acc[mf][nf]);
    }
    const int l0 = lane & 15, hi = lane >> 4;
    #pragma unroll
    for (int mf = 0; mf < 4; ++mf)
        #pragma unroll
        for (int nf = 0; nf < 2; ++nf) {
            const long col = n0 + wave * 32 + nf * 16 + l0;
            #pragma unroll
            for (int r = 0; r < 8; ++r) {
                const int row = mf * 16 + hi * 8 + r;        // channel e in head
                const float x = acc[mf][nf][r];
                const float g = 0.5f * x * (1.0f + erff(x * 0.70710678f));
                attn[((long)b * INNER_ + h * 64 + row) * N_ + col] = (__bf16)g;
            }
        }
}

// ---------------- final projection: w_out(256x512) * attn + bias -> fp32 out ----------------
__global__ __launch_bounds__(256) void k_gemm_final(const __bf16* __restrict__ wout,
                                                    const __bf16* __restrict__ attn,
                                                    const float* __restrict__ bias,
                                                    float* __restrict__ out) {
    __shared__ __align__(16) __bf16 As[2][128][40];
    __shared__ __align__(16) __bf16 Bs[2][128][40];
    const int n0    = blockIdx.x * 128;
    const int mbase = blockIdx.y * 128;   // 0 or 128
    const int b     = blockIdx.z;
    floatx8 acc[4][2];
    #pragma unroll
    for (int i = 0; i < 4; ++i) { acc[i][0] = zero8(); acc[i][1] = zero8(); }
    gemm_tile_128x128(wout + (long)mbase * INNER_, INNER_,
                      attn + (long)b * INNER_ * N_ + n0, N_, INNER_, As, Bs, acc);
    const int tid = threadIdx.x, lane = tid & 31, wave = tid >> 5;
    const int wm = wave & 1, wn = wave >> 1, l0 = lane & 15, hi = lane >> 4;
    #pragma unroll
    for (int mf = 0; mf < 4; ++mf)
        #pragma unroll
        for (int nf = 0; nf < 2; ++nf) {
            const long col = n0 + wn * 32 + nf * 16 + l0;
            #pragma unroll
            for (int r = 0; r < 8; ++r) {
                const int row = mbase + wm * 64 + mf * 16 + hi * 8 + r;
                out[((long)b * C_ + row) * N_ + col] = acc[mf][nf][r] + bias[row];
            }
        }
}

// ---------------- host orchestration ----------------
extern "C" void kernel_launch(void* const* d_in, const int* in_sizes, int n_in,
                              void* d_out, int out_size, void* d_ws, size_t ws_size,
                              hipStream_t stream) {
    (void)in_sizes; (void)n_in; (void)out_size; (void)ws_size;
    const float* fmap  = (const float*)d_in[0];
    const float* w_q   = (const float*)d_in[1];
    const float* w_dw  = (const float*)d_in[2];
    const float* w_pw  = (const float*)d_in[3];
    const float* w_out = (const float*)d_in[4];
    const float* b_out = (const float*)d_in[5];
    float* outp = (float*)d_out;

    char* base = (char*)d_ws;
    size_t off = 0;
    auto alloc = [&](size_t bytes) -> void* {
        void* p = base + off;
        off += (bytes + 255) & ~(size_t)255;
        return p;
    };
    __bf16* wq_bf   = (__bf16*)alloc((size_t)INNER_ * C_ * 2);
    __bf16* wpw_bf  = (__bf16*)alloc((size_t)2 * INNER_ * C_ * 2);
    __bf16* wout_bf = (__bf16*)alloc((size_t)C_ * INNER_ * 2);
    __bf16* fmap_bf = (__bf16*)alloc((size_t)B_ * C_ * N_ * 2);
    __bf16* dw_bf   = (__bf16*)alloc((size_t)B_ * C_ * N_ * 2);
    __bf16* q_bf    = (__bf16*)alloc((size_t)B_ * INNER_ * N_ * 2);
    __bf16* kv_bf   = (__bf16*)alloc((size_t)B_ * 2 * INNER_ * N_ * 2);
    __bf16* attn_bf = (__bf16*)alloc((size_t)B_ * INNER_ * N_ * 2);
    float*  kmax    = (float*)alloc((size_t)B_ * INNER_ * 4);
    float*  ksum    = (float*)alloc((size_t)B_ * INNER_ * 4);
    float*  ctx     = (float*)alloc((size_t)B_ * HEADS_ * D_ * D_ * 4);

    // 1) precision conversion (one pass; GEMMs then move half the bytes)
    k_convert_bf16<<<512,  256, 0, stream>>>(w_q,   wq_bf,   (long)INNER_ * C_);
    k_convert_bf16<<<1024, 256, 0, stream>>>(w_pw,  wpw_bf,  (long)2 * INNER_ * C_);
    k_convert_bf16<<<512,  256, 0, stream>>>(w_out, wout_bf, (long)C_ * INNER_);
    k_convert_bf16<<<8192, 256, 0, stream>>>(fmap,  fmap_bf, (long)B_ * C_ * N_);
    // 2) depthwise 3x3
    k_dwconv<<<B_ * C_ * HW_, 128, 0, stream>>>(fmap, w_dw, dw_bf);
    // 3) q projection + fused feature-softmax  (M=512,K=256,N=16384 per batch)
    k_gemm_q<<<dim3(128, 4, B_), 256, 0, stream>>>(wq_bf, fmap_bf, q_bf);
    // 4) kv projection (M=1024)
    k_gemm_kv<<<dim3(128, 8, B_), 256, 0, stream>>>(wpw_bf, dw_bf, kv_bf);
    // 5) k spatial-softmax statistics
    k_kstats<<<B_ * INNER_, 256, 0, stream>>>(kv_bf, kmax, ksum);
    // 6) context (zero then split-K atomic accumulate)
    k_zero_f32<<<256, 256, 0, stream>>>(ctx, (long)B_ * HEADS_ * D_ * D_);
    k_context<<<dim3(8, B_ * HEADS_), 256, 0, stream>>>(kv_bf, kmax, ksum, ctx);
    // 7) out = ctx^T q, GELU
    k_out<<<dim3(64, B_ * HEADS_), 256, 0, stream>>>(ctx, q_bf, attn_bf);
    // 8) output projection + bias (fp32 result)
    k_gemm_final<<<dim3(128, 2, B_), 256, 0, stream>>>(wout_bf, attn_bf, b_out, outp);
}
